// PatchcoreRS_69552700391708
// MI455X (gfx1250) — compile-verified, matrix operand order checked by reading, not run on hardware
//
#include <hip/hip_runtime.h>
#include <hip/hip_bf16.h>

// Problem constants (fixed by the reference).
#define N_EMB 25088
#define M_BANK 16384
#define D_DIM 1536
#define KTOP 9

// Tiling.
#define BLK_E 64                    // embedding rows per block
#define BLK_M 128                   // memory-bank rows per tile iteration
#define M_CHUNK 2048                // memory-bank rows per block (looped)
#define N_CHUNK (M_BANK / M_CHUNK)  // 8 chunks along M
#define M_ITERS (M_CHUNK / BLK_M)   // 16
#define KSTEP 32                    // K per bf16 WMMA
#define K_ITERS (D_DIM / KSTEP)     // 48

typedef __attribute__((ext_vector_type(16))) __bf16 bf16x16;
typedef __attribute__((ext_vector_type(4)))  __bf16 bf16x4;
typedef __attribute__((ext_vector_type(8)))  float  f32x8;
typedef __attribute__((ext_vector_type(4)))  float  f32x4;

static __device__ __forceinline__ __bf16 f2bf(float x) {
  unsigned u = __builtin_bit_cast(unsigned, x);
  u += 0x7FFFu + ((u >> 16) & 1u);            // round-to-nearest-even
  unsigned short h = (unsigned short)(u >> 16);
  return __builtin_bit_cast(__bf16, h);
}
static __device__ __forceinline__ float bf2f(__bf16 b) {
  unsigned u = ((unsigned)__builtin_bit_cast(unsigned short, b)) << 16;
  return __builtin_bit_cast(float, u);
}

// Flat LDS addresses are {SHARED_BASE, offset[31:0]} -> low 32 bits = LDS offset.
static __device__ __forceinline__ unsigned lds_off(const void* p) {
  return (unsigned)(unsigned long long)p;
}

// One async B128 copy global -> LDS (tracked by ASYNCcnt). GVS addressing:
// mem = SGPR64 base + VGPR32 byte offset.
#define ASYNC_B128(ldsoff, base64, voff)                                   \
  asm volatile("global_load_async_to_lds_b128 %0, %1, %2"                  \
               :: "v"(ldsoff), "v"(voff), "s"(base64) : "memory")

// Load one 16-bit A/B fragment from LDS per the CDNA5 16x32 layout:
// lane holds 8 bf16 at (row*32 + khalf) and 8 bf16 at (row*32 + khalf + 16).
static __device__ __forceinline__ bf16x16 load_frag(const __bf16* p) {
  union U { f32x4 q[2]; bf16x16 v; } u;
  u.q[0] = *(const f32x4*)(p);
  u.q[1] = *(const f32x4*)(p + 16);
  return u.v;
}

// Unrolled sorted-insert of d into ascending best[KTOP]; stays in VGPRs.
#define TOPK_INSERT(best, d)                                         \
  do {                                                               \
    if ((d) < (best)[KTOP - 1]) {                                    \
      (best)[KTOP - 1] = (d);                                        \
      _Pragma("unroll")                                              \
      for (int _s = KTOP - 1; _s > 0; --_s) {                        \
        float _lo = fminf((best)[_s - 1], (best)[_s]);               \
        float _hi = fmaxf((best)[_s - 1], (best)[_s]);               \
        (best)[_s - 1] = _lo; (best)[_s] = _hi;                      \
      }                                                              \
    }                                                                \
  } while (0)

// ---------------------------------------------------------------------------
// Kernel 0: row norms for embedding (x) and memory bank (y). One wave per row.
// ---------------------------------------------------------------------------
__global__ __launch_bounds__(256)
void patchcore_norms(const float* __restrict__ emb,
                     const float* __restrict__ bank,
                     float* __restrict__ xnorm,
                     float* __restrict__ ynorm) {
  const int row  = blockIdx.x * 8 + (threadIdx.x >> 5);
  const int lane = threadIdx.x & 31;
  const float* src;
  float* dst;
  if (row < N_EMB) {
    src = emb + (size_t)row * D_DIM;
    dst = xnorm + row;
  } else if (row < N_EMB + M_BANK) {
    src = bank + (size_t)(row - N_EMB) * D_DIM;
    dst = ynorm + (row - N_EMB);
  } else {
    return;
  }
  float s = 0.0f;
  for (int k = lane; k < D_DIM; k += 32) {
    float v = src[k];
    s = fmaf(v, v, s);
  }
#pragma unroll
  for (int off = 16; off > 0; off >>= 1) s += __shfl_xor(s, off, 32);
  if (lane == 0) *dst = s;
}

// ---------------------------------------------------------------------------
// Kernel 0b: one-shot fp32 -> bf16 hi/lo split (the split is input-only; do
// it once instead of once per tile load).
// ---------------------------------------------------------------------------
__global__ __launch_bounds__(256)
void convert_hilo(const float* __restrict__ src,
                  __bf16* __restrict__ hi, __bf16* __restrict__ lo, int n4) {
  const int i = blockIdx.x * 256 + threadIdx.x;
  if (i >= n4) return;
  f32x4 v = ((const f32x4*)src)[i];
  bf16x4 h, l;
#pragma unroll
  for (int c = 0; c < 4; ++c) {
    __bf16 hh = f2bf(v[c]);
    h[c] = hh;
    l[c] = f2bf(v[c] - bf2f(hh));
  }
  ((bf16x4*)hi)[i] = h;
  ((bf16x4*)lo)[i] = l;
}

// ---------------------------------------------------------------------------
// Kernel 1 (fast path): bf16x3 WMMA GEMM with double-buffered async-to-LDS
// tile copies + fused distance/top-9. Grid (N_EMB/BLK_E, N_CHUNK), 256 thr.
// ---------------------------------------------------------------------------
__global__ __launch_bounds__(256)
void patchcore_gemm_topk_async(const __bf16* __restrict__ embHi,
                               const __bf16* __restrict__ embLo,
                               const __bf16* __restrict__ bankHi,
                               const __bf16* __restrict__ bankLo,
                               const float* __restrict__ xnorm,
                               const float* __restrict__ ynorm,
                               float* __restrict__ partial) {
  __shared__ __bf16 sAhi[2][BLK_M * KSTEP];   // 2 x 8 KB
  __shared__ __bf16 sAlo[2][BLK_M * KSTEP];   // 2 x 8 KB
  __shared__ __bf16 sBhi[2][BLK_E * KSTEP];   // 2 x 4 KB
  __shared__ __bf16 sBlo[2][BLK_E * KSTEP];   // 2 x 4 KB
  __shared__ float  sD[BLK_E * BLK_M];        // 32 KB
  __shared__ float  sM[256 * KTOP];           //  9 KB

  const int tid    = threadIdx.x;
  const int wave   = tid >> 5;
  const int lane   = tid & 31;
  const int nBase  = blockIdx.x * BLK_E;
  const int mChunk = blockIdx.y * M_CHUNK;

  const int eRow = tid >> 2;
  const int qSeg = tid & 3;
  const float xnv = xnorm[nBase + eRow];

  float best[KTOP];
#pragma unroll
  for (int s = 0; s < KTOP; ++s) best[s] = 3.0e38f;

  const int mrow  = wave * 16 + (lane & 15);
  const int khalf = (lane < 16) ? 0 : 8;

  // Async copy assignment: A tile = 128 rows x 64 B (512 B128 chunks -> 2/thr
  // per plane), B tile = 64 rows x 64 B (256 chunks -> 1/thr per plane).
  const int rA0 = tid >> 2;                       // 0..63
  const unsigned sg16 = (unsigned)(tid & 3) * 16u;
  const unsigned aoff0 = (unsigned)rA0 * 64u + sg16;
  const unsigned aoff1 = aoff0 + 64u * 64u;       // rows 64..127
  const unsigned boff  = (unsigned)rA0 * 64u + sg16;

  const unsigned oAhi[2] = { lds_off(&sAhi[0][0]), lds_off(&sAhi[1][0]) };
  const unsigned oAlo[2] = { lds_off(&sAlo[0][0]), lds_off(&sAlo[1][0]) };
  const unsigned oBhi[2] = { lds_off(&sBhi[0][0]), lds_off(&sBhi[1][0]) };
  const unsigned oBlo[2] = { lds_off(&sBlo[0][0]), lds_off(&sBlo[1][0]) };

  const unsigned long long gAh = (unsigned long long)bankHi;
  const unsigned long long gAl = (unsigned long long)bankLo;
  const unsigned long long gBh = (unsigned long long)embHi;
  const unsigned long long gBl = (unsigned long long)embLo;

  const unsigned rowBytes = (unsigned)(D_DIM * 2);
  const unsigned gB0 = (unsigned)(nBase + rA0) * rowBytes + sg16;

  for (int it = 0; it < M_ITERS; ++it) {
    const int mBase = mChunk + it * BLK_M;
    const unsigned gA0 = (unsigned)(mBase + rA0) * rowBytes + sg16;
    const unsigned gA1 = gA0 + 64u * rowBytes;

    f32x8 acc[4];
#pragma unroll
    for (int t = 0; t < 4; ++t)
#pragma unroll
      for (int r = 0; r < 8; ++r) acc[t][r] = 0.0f;

    // Prologue: stage K-step 0 into buffer 0.
    ASYNC_B128(oAhi[0] + aoff0, gAh, gA0);
    ASYNC_B128(oAhi[0] + aoff1, gAh, gA1);
    ASYNC_B128(oAlo[0] + aoff0, gAl, gA0);
    ASYNC_B128(oAlo[0] + aoff1, gAl, gA1);
    ASYNC_B128(oBhi[0] + boff,  gBh, gB0);
    ASYNC_B128(oBlo[0] + boff,  gBl, gB0);

    for (int kk = 0; kk < K_ITERS; ++kk) {
      const int cur = kk & 1;
      if (kk + 1 < K_ITERS) {
        // Prefetch next K-step into the other buffer, then wait for current
        // buffer only (own 6 newest ops may remain outstanding).
        const int nb = cur ^ 1;
        const unsigned kb = (unsigned)(kk + 1) * (KSTEP * 2);
        ASYNC_B128(oAhi[nb] + aoff0, gAh, gA0 + kb);
        ASYNC_B128(oAhi[nb] + aoff1, gAh, gA1 + kb);
        ASYNC_B128(oAlo[nb] + aoff0, gAl, gA0 + kb);
        ASYNC_B128(oAlo[nb] + aoff1, gAl, gA1 + kb);
        ASYNC_B128(oBhi[nb] + boff,  gBh, gB0 + kb);
        ASYNC_B128(oBlo[nb] + boff,  gBl, gB0 + kb);
        asm volatile("s_wait_asynccnt 0x6" ::: "memory");
      } else {
        asm volatile("s_wait_asynccnt 0x0" ::: "memory");
      }
      __syncthreads();  // all waves' copies into buffer `cur` complete

      bf16x16 a_hi = load_frag(&sAhi[cur][mrow * KSTEP + khalf]);
      bf16x16 a_lo = load_frag(&sAlo[cur][mrow * KSTEP + khalf]);
#pragma unroll
      for (int t = 0; t < 4; ++t) {
        const int ecol = t * 16 + (lane & 15);
        bf16x16 b_hi = load_frag(&sBhi[cur][ecol * KSTEP + khalf]);
        bf16x16 b_lo = load_frag(&sBlo[cur][ecol * KSTEP + khalf]);
        acc[t] = __builtin_amdgcn_wmma_f32_16x16x32_bf16(
            false, a_hi, false, b_hi, (short)0, acc[t], false, false);
        acc[t] = __builtin_amdgcn_wmma_f32_16x16x32_bf16(
            false, a_lo, false, b_hi, (short)0, acc[t], false, false);
        acc[t] = __builtin_amdgcn_wmma_f32_16x16x32_bf16(
            false, a_hi, false, b_lo, (short)0, acc[t], false, false);
      }
      __syncthreads();  // all waves done reading buffer `cur` before reuse
    }

    // Scatter accumulators to sD[e][m] per the 16x16 f32 C/D layout.
    const int moff = wave * 16 + ((lane >> 4) << 3);
#pragma unroll
    for (int t = 0; t < 4; ++t) {
      const int e = t * 16 + (lane & 15);
#pragma unroll
      for (int r = 0; r < 8; ++r)
        sD[e * BLK_M + moff + r] = acc[t][r];
    }
    __syncthreads();

    // Fused epilogue: distance + running top-9.
#pragma unroll 4
    for (int j = 0; j < 32; ++j) {
      const int m = qSeg * 32 + j;
      float dot = sD[eRow * BLK_M + m];
      float d = fmaf(-2.0f, dot, xnv + ynorm[mBase + m]);
      d = __builtin_sqrtf(fmaxf(d, 1e-12f));
      TOPK_INSERT(best, d);
    }
    __syncthreads();  // sD reads complete before next iteration's writes
  }

  // Merge the 4 per-quarter lists of each embedding row.
#pragma unroll
  for (int s = 0; s < KTOP; ++s) sM[tid * KTOP + s] = best[s];
  __syncthreads();
  if (qSeg == 0) {
#pragma unroll
    for (int g = 1; g < 4; ++g) {
      const float* o = sM + (tid + g) * KTOP;
#pragma unroll
      for (int s = 0; s < KTOP; ++s) {
        float d = o[s];
        TOPK_INSERT(best, d);
      }
    }
    float* pp = partial + (size_t)(nBase + eRow) * (N_CHUNK * KTOP)
              + (size_t)blockIdx.y * KTOP;
#pragma unroll
    for (int s = 0; s < KTOP; ++s) pp[s] = best[s];
  }
}

// ---------------------------------------------------------------------------
// Kernel 1 (fallback, small workspace): on-the-fly hi/lo conversion in LDS.
// ---------------------------------------------------------------------------
__global__ __launch_bounds__(256)
void patchcore_gemm_topk(const float* __restrict__ emb,
                         const float* __restrict__ bank,
                         const float* __restrict__ xnorm,
                         const float* __restrict__ ynorm,
                         float* __restrict__ partial) {
  __shared__ __bf16 sA_hi[BLK_M * KSTEP];
  __shared__ __bf16 sA_lo[BLK_M * KSTEP];
  __shared__ __bf16 sB_hi[BLK_E * KSTEP];
  __shared__ __bf16 sB_lo[BLK_E * KSTEP];
  __shared__ float  sD[BLK_E * BLK_M];
  __shared__ float  sM[256 * KTOP];

  const int tid    = threadIdx.x;
  const int wave   = tid >> 5;
  const int lane   = tid & 31;
  const int nBase  = blockIdx.x * BLK_E;
  const int mChunk = blockIdx.y * M_CHUNK;

  const int eRow = tid >> 2;
  const int qSeg = tid & 3;
  const float xnv = xnorm[nBase + eRow];

  float best[KTOP];
#pragma unroll
  for (int s = 0; s < KTOP; ++s) best[s] = 3.0e38f;

  const int mrow  = wave * 16 + (lane & 15);
  const int khalf = (lane < 16) ? 0 : 8;

  for (int it = 0; it < M_ITERS; ++it) {
    const int mBase = mChunk + it * BLK_M;

    f32x8 acc[4];
#pragma unroll
    for (int t = 0; t < 4; ++t)
#pragma unroll
      for (int r = 0; r < 8; ++r) acc[t][r] = 0.0f;

    for (int kk = 0; kk < K_ITERS; ++kk) {
      const int koff = kk * KSTEP;
      __syncthreads();
#pragma unroll
      for (int i = 0; i < 4; ++i) {
        int fi  = tid + i * 256;
        int row = fi >> 3;
        int col = (fi & 7) << 2;
        f32x4 v = *(const f32x4*)(bank + (size_t)(mBase + row) * D_DIM + koff + col);
#pragma unroll
        for (int c = 0; c < 4; ++c) {
          __bf16 hi = f2bf(v[c]);
          sA_hi[row * KSTEP + col + c] = hi;
          sA_lo[row * KSTEP + col + c] = f2bf(v[c] - bf2f(hi));
        }
      }
#pragma unroll
      for (int i = 0; i < 2; ++i) {
        int fi  = tid + i * 256;
        int row = fi >> 3;
        int col = (fi & 7) << 2;
        f32x4 v = *(const f32x4*)(emb + (size_t)(nBase + row) * D_DIM + koff + col);
#pragma unroll
        for (int c = 0; c < 4; ++c) {
          __bf16 hi = f2bf(v[c]);
          sB_hi[row * KSTEP + col + c] = hi;
          sB_lo[row * KSTEP + col + c] = f2bf(v[c] - bf2f(hi));
        }
      }
      __syncthreads();

      bf16x16 a_hi = load_frag(sA_hi + mrow * KSTEP + khalf);
      bf16x16 a_lo = load_frag(sA_lo + mrow * KSTEP + khalf);
#pragma unroll
      for (int t = 0; t < 4; ++t) {
        const int ecol = t * 16 + (lane & 15);
        bf16x16 b_hi = load_frag(sB_hi + ecol * KSTEP + khalf);
        bf16x16 b_lo = load_frag(sB_lo + ecol * KSTEP + khalf);
        acc[t] = __builtin_amdgcn_wmma_f32_16x16x32_bf16(
            false, a_hi, false, b_hi, (short)0, acc[t], false, false);
        acc[t] = __builtin_amdgcn_wmma_f32_16x16x32_bf16(
            false, a_lo, false, b_hi, (short)0, acc[t], false, false);
        acc[t] = __builtin_amdgcn_wmma_f32_16x16x32_bf16(
            false, a_hi, false, b_lo, (short)0, acc[t], false, false);
      }
    }
    __syncthreads();

    const int moff = wave * 16 + ((lane >> 4) << 3);
#pragma unroll
    for (int t = 0; t < 4; ++t) {
      const int e = t * 16 + (lane & 15);
#pragma unroll
      for (int r = 0; r < 8; ++r)
        sD[e * BLK_M + moff + r] = acc[t][r];
    }
    __syncthreads();

#pragma unroll 4
    for (int j = 0; j < 32; ++j) {
      const int m = qSeg * 32 + j;
      float dot = sD[eRow * BLK_M + m];
      float d = fmaf(-2.0f, dot, xnv + ynorm[mBase + m]);
      d = __builtin_sqrtf(fmaxf(d, 1e-12f));
      TOPK_INSERT(best, d);
    }
  }

#pragma unroll
  for (int s = 0; s < KTOP; ++s) sM[tid * KTOP + s] = best[s];
  __syncthreads();
  if (qSeg == 0) {
#pragma unroll
    for (int g = 1; g < 4; ++g) {
      const float* o = sM + (tid + g) * KTOP;
#pragma unroll
      for (int s = 0; s < KTOP; ++s) {
        float d = o[s];
        TOPK_INSERT(best, d);
      }
    }
    float* pp = partial + (size_t)(nBase + eRow) * (N_CHUNK * KTOP)
              + (size_t)blockIdx.y * KTOP;
#pragma unroll
    for (int s = 0; s < KTOP; ++s) pp[s] = best[s];
  }
}

// ---------------------------------------------------------------------------
// Kernel 2: merge N_CHUNK partial top-9 lists -> final top-9 (ascending).
// ---------------------------------------------------------------------------
__global__ __launch_bounds__(256)
void patchcore_merge(const float* __restrict__ partial,
                     float* __restrict__ out) {
  const int row = blockIdx.x * 256 + threadIdx.x;
  if (row >= N_EMB) return;
  float best[KTOP];
#pragma unroll
  for (int s = 0; s < KTOP; ++s) best[s] = 3.0e38f;
  const float* p = partial + (size_t)row * (N_CHUNK * KTOP);
  for (int i = 0; i < N_CHUNK * KTOP; ++i) {
    float d = p[i];
    TOPK_INSERT(best, d);
  }
#pragma unroll
  for (int s = 0; s < KTOP; ++s) out[(size_t)row * KTOP + s] = best[s];
}

// ---------------------------------------------------------------------------
extern "C" void kernel_launch(void* const* d_in, const int* in_sizes, int n_in,
                              void* d_out, int out_size, void* d_ws, size_t ws_size,
                              hipStream_t stream) {
  (void)in_sizes; (void)n_in; (void)out_size;
  const float* emb  = (const float*)d_in[0];   // [N_EMB, D_DIM]
  const float* bank = (const float*)d_in[1];   // [M_BANK, D_DIM]

  float* xnorm   = (float*)d_ws;
  float* ynorm   = xnorm + N_EMB;
  float* partial = ynorm + M_BANK;
  float* out     = (float*)d_out;

  const size_t baseBytes = ((size_t)N_EMB + M_BANK
                            + (size_t)N_EMB * N_CHUNK * KTOP) * sizeof(float);
  const size_t embPlane  = (size_t)N_EMB * D_DIM;    // bf16 elements
  const size_t bankPlane = (size_t)M_BANK * D_DIM;
  const size_t fastBytes = baseBytes + 2 * (embPlane + bankPlane) * sizeof(__bf16);

  const int rows = N_EMB + M_BANK;
  patchcore_norms<<<(rows + 7) / 8, 256, 0, stream>>>(emb, bank, xnorm, ynorm);

  dim3 grid(N_EMB / BLK_E, N_CHUNK);               // 392 x 8 blocks

  if (ws_size >= fastBytes) {
    __bf16* embHi  = (__bf16*)((char*)d_ws + baseBytes);
    __bf16* embLo  = embHi + embPlane;
    __bf16* bankHi = embLo + embPlane;
    __bf16* bankLo = bankHi + bankPlane;

    const int n4e = (int)(embPlane / 4);
    const int n4b = (int)(bankPlane / 4);
    convert_hilo<<<(n4e + 255) / 256, 256, 0, stream>>>(emb, embHi, embLo, n4e);
    convert_hilo<<<(n4b + 255) / 256, 256, 0, stream>>>(bank, bankHi, bankLo, n4b);

    patchcore_gemm_topk_async<<<grid, 256, 0, stream>>>(
        embHi, embLo, bankHi, bankLo, xnorm, ynorm, partial);
  } else {
    patchcore_gemm_topk<<<grid, 256, 0, stream>>>(emb, bank, xnorm, ynorm, partial);
  }

  patchcore_merge<<<(N_EMB + 255) / 256, 256, 0, stream>>>(partial, out);
}
